// MultiHeadSelfAttention_32229434589604
// MI455X (gfx1250) — matrix operand order repaired and checked by reading, not compiled, not run on hardware
//
#include <hip/hip_runtime.h>
#include <math.h>

// ---------------------------------------------------------------------------
// MHA forward, CDNA5 (gfx1250, wave32) WMMA f16 path with fp32 accumulate.
// B=2, S=2048, E=1024, H=16, D=64.
// ---------------------------------------------------------------------------

typedef _Float16 v16h __attribute__((ext_vector_type(16)));
typedef _Float16 v8h  __attribute__((ext_vector_type(8)));
typedef float    v8f  __attribute__((ext_vector_type(8)));

#define S_LEN   2048
#define E_DIM   1024
#define N_HEAD  16
#define H_DIM   64
#define NTOK    4096            // B * S

__device__ __forceinline__ v8f wmma_f16(v16h a, v16h b, v8f c) {
    // v_wmma_f32_16x16x32_f16  D = A*B + C
    return __builtin_amdgcn_wmma_f32_16x16x32_f16(
        false, a, false, b, (short)0, c, false, false);
}

// A-fragment, 16x32 f16 (M x K), row-major source with leading dim lda.
// Lane l (l<16): row M=l, holds K = {k..k+7, k+16..k+23}
// Lane l (l>=16): row M=l-16, holds K = {k+8..k+15, k+24..k+31}
__device__ __forceinline__ v16h load_a_frag(const _Float16* __restrict__ A,
                                            int lda, int row_base, int k, int lane) {
    int m  = row_base + (lane & 15);
    int kb = k + ((lane & 16) ? 8 : 0);
    const _Float16* p = A + (size_t)m * lda + kb;
    v8h lo = *(const v8h*)p;
    v8h hi = *(const v8h*)(p + 16);
    v16h r;
#pragma unroll
    for (int i = 0; i < 8; ++i) { r[i] = lo[i]; r[i + 8] = hi[i]; }
    return r;
}

// B-fragment, 32x16 f16 (K x N), built from ROWS of a row-major matrix W:
// B[k'][n] = W[col_base+n][k']  (i.e. NT gemm, per-lane contiguous in K).
// Lane l (l<16): col N=l, K = {k..k+7, k+16..k+23} ; lane l>=16: col N=l-16, K = {k+8..k+15, k+24..k+31}
__device__ __forceinline__ v16h load_bT_frag(const _Float16* __restrict__ W,
                                             int ldw, int col_base, int k, int lane) {
    int n  = col_base + (lane & 15);
    int kb = k + ((lane & 16) ? 8 : 0);
    const _Float16* p = W + (size_t)n * ldw + kb;
    v8h lo = *(const v8h*)p;
    v8h hi = *(const v8h*)(p + 16);
    v16h r;
#pragma unroll
    for (int i = 0; i < 8; ++i) { r[i] = lo[i]; r[i + 8] = hi[i]; }
    return r;
}

// ---------------------------------------------------------------------------
// fp32 -> f16 conversion
// ---------------------------------------------------------------------------
__global__ void cvt_f32_f16(const float* __restrict__ src,
                            _Float16* __restrict__ dst, int n) {
    int i  = blockIdx.x * blockDim.x + threadIdx.x;
    int st = gridDim.x * blockDim.x;
    for (; i < n; i += st) dst[i] = (_Float16)src[i];
}

// ---------------------------------------------------------------------------
// GEMM1: qkv[m,f] = X[m,:] . Wqkv[f,:]  (M=4096, N=3072, K=1024)
// Each wave: 32x32 output tile. Scatter to Q[b,h,s,d]*0.125, K[b,h,s,d],
// Vt[b,h,d,s] as f16.
// ---------------------------------------------------------------------------
__global__ void __launch_bounds__(128)
gemm_qkv(const _Float16* __restrict__ Xh, const _Float16* __restrict__ Wh,
         _Float16* __restrict__ Qh, _Float16* __restrict__ Kh,
         _Float16* __restrict__ Vt) {
    const int lane = threadIdx.x & 31;
    const int wave = threadIdx.x >> 5;
    const int tile = blockIdx.x * 4 + wave;          // 128 x 96 = 12288 tiles
    const int mt = (tile / 96) * 32;
    const int ft = (tile % 96) * 32;

    v8f c00 = {}, c01 = {}, c10 = {}, c11 = {};
#pragma unroll 2
    for (int k = 0; k < E_DIM; k += 32) {
        v16h a0 = load_a_frag(Xh, E_DIM, mt,      k, lane);
        v16h a1 = load_a_frag(Xh, E_DIM, mt + 16, k, lane);
        v16h b0 = load_bT_frag(Wh, E_DIM, ft,      k, lane);
        v16h b1 = load_bT_frag(Wh, E_DIM, ft + 16, k, lane);
        c00 = wmma_f16(a0, b0, c00);
        c01 = wmma_f16(a0, b1, c01);
        c10 = wmma_f16(a1, b0, c10);
        c11 = wmma_f16(a1, b1, c11);
    }

    const int half8 = (lane & 16) ? 8 : 0;
#pragma unroll
    for (int mi = 0; mi < 2; ++mi) {
#pragma unroll
        for (int ni = 0; ni < 2; ++ni) {
            v8f c = mi ? (ni ? c11 : c10) : (ni ? c01 : c00);
            int f    = ft + ni * 16 + (lane & 15);
            int part = f >> 10;            // 0=Q 1=K 2=V
            int feat = f & 1023;
            int h    = feat >> 6;
            int d    = feat & 63;
#pragma unroll
            for (int i = 0; i < 8; ++i) {
                int r = mt + mi * 16 + half8 + i;     // token index
                int b = r >> 11;
                int s = r & (S_LEN - 1);
                float v = c[i];
                size_t bh = (size_t)b * N_HEAD + h;
                if (part == 0)
                    Qh[(bh * S_LEN + s) * H_DIM + d] = (_Float16)(v * 0.125f);
                else if (part == 1)
                    Kh[(bh * S_LEN + s) * H_DIM + d] = (_Float16)v;
                else
                    Vt[(bh * H_DIM + d) * S_LEN + s] = (_Float16)v;
            }
        }
    }
}

// ---------------------------------------------------------------------------
// Flash attention: one wave = 16 query rows of one (b,h); stream keys in
// 32-wide steps with online softmax; O accumulated as 4 f32 WMMA tiles.
// ---------------------------------------------------------------------------
__global__ void __launch_bounds__(128)
attn_flash(const _Float16* __restrict__ Qh, const _Float16* __restrict__ Kh,
           const _Float16* __restrict__ Vt, _Float16* __restrict__ Eh) {
    __shared__ _Float16 plds[4][16][40];   // per-wave P tile, padded rows (80B)

    const int lane  = threadIdx.x & 31;
    const int wave  = threadIdx.x >> 5;
    const int bh    = blockIdx.y;                   // 0..31
    const int qbase = blockIdx.x * 64 + wave * 16;  // query block of this wave
    const int half8 = (lane & 16) ? 8 : 0;

    const _Float16* Qp = Qh + (size_t)bh * S_LEN * H_DIM;
    const _Float16* Kp = Kh + (size_t)bh * S_LEN * H_DIM;
    const _Float16* Vp = Vt + (size_t)bh * H_DIM * S_LEN;

    // Q fragments (invariant over the key loop): 16 x 64 split into two K=32
    v16h aq0 = load_a_frag(Qp, H_DIM, qbase, 0,  lane);
    v16h aq1 = load_a_frag(Qp, H_DIM, qbase, 32, lane);

    v8f o0 = {}, o1 = {}, o2 = {}, o3 = {};
    float mrow[8], lrow[8];
#pragma unroll
    for (int i = 0; i < 8; ++i) { mrow[i] = -1e30f; lrow[i] = 0.0f; }

    const int nsteps = (qbase + 15) / 32 + 1;       // causal bound
    for (int ks = 0; ks < nsteps; ++ks) {
        const int kb = ks * 32;

        // scores: 16 queries x 32 keys = two 16x16 C tiles
        v8f s0 = {}, s1 = {};
        s0 = wmma_f16(aq0, load_bT_frag(Kp, H_DIM, kb,      0,  lane), s0);
        s0 = wmma_f16(aq1, load_bT_frag(Kp, H_DIM, kb,      32, lane), s0);
        s1 = wmma_f16(aq0, load_bT_frag(Kp, H_DIM, kb + 16, 0,  lane), s1);
        s1 = wmma_f16(aq1, load_bT_frag(Kp, H_DIM, kb + 16, 32, lane), s1);

        const int key0 = kb + (lane & 15);
        const int key1 = key0 + 16;
#pragma unroll
        for (int i = 0; i < 8; ++i) {
            int r = qbase + half8 + i;
            float x0 = (key0 <= r) ? s0[i] : -1e30f;   // causal mask
            float x1 = (key1 <= r) ? s1[i] : -1e30f;
            // row max across the 16 lanes holding this row
            float t = fmaxf(x0, x1);
#pragma unroll
            for (int off = 1; off < 16; off <<= 1)
                t = fmaxf(t, __shfl_xor(t, off, 16));
            float mn    = fmaxf(mrow[i], t);
            float alpha = __expf(mrow[i] - mn);
            float p0    = __expf(x0 - mn);
            float p1    = __expf(x1 - mn);
            float rs    = p0 + p1;
#pragma unroll
            for (int off = 1; off < 16; off <<= 1)
                rs += __shfl_xor(rs, off, 16);
            lrow[i] = lrow[i] * alpha + rs;
            mrow[i] = mn;
            o0[i] *= alpha; o1[i] *= alpha; o2[i] *= alpha; o3[i] *= alpha;
            // stash P (f16) in C-layout positions
            plds[wave][half8 + i][lane & 15]        = (_Float16)p0;
            plds[wave][half8 + i][16 + (lane & 15)] = (_Float16)p1;
        }
        asm volatile("s_wait_dscnt 0" ::: "memory");

        // re-read P as an A fragment (16x32)
        {
            int m  = lane & 15;
            int kp = (lane & 16) ? 8 : 0;
            const _Float16* pr = &plds[wave][m][kp];
            v8h lo = *(const v8h*)pr;
            v8h hi = *(const v8h*)(pr + 16);
            v16h ap;
#pragma unroll
            for (int i = 0; i < 8; ++i) { ap[i] = lo[i]; ap[i + 8] = hi[i]; }

            // O += P @ V  (V transposed in memory -> contiguous B frags)
            o0 = wmma_f16(ap, load_bT_frag(Vp, S_LEN,  0, kb, lane), o0);
            o1 = wmma_f16(ap, load_bT_frag(Vp, S_LEN, 16, kb, lane), o1);
            o2 = wmma_f16(ap, load_bT_frag(Vp, S_LEN, 32, kb, lane), o2);
            o3 = wmma_f16(ap, load_bT_frag(Vp, S_LEN, 48, kb, lane), o3);
        }
    }

    // epilogue: O /= l, write E[b, s, h*64 + d] (f16)
    const int b = bh >> 4;
    const int h = bh & 15;
#pragma unroll
    for (int i = 0; i < 8; ++i) {
        float inv = 1.0f / lrow[i];
        int s = qbase + half8 + i;
        size_t row = ((size_t)b * S_LEN + s) * E_DIM + h * H_DIM;
        Eh[row +  0 + (lane & 15)] = (_Float16)(o0[i] * inv);
        Eh[row + 16 + (lane & 15)] = (_Float16)(o1[i] * inv);
        Eh[row + 32 + (lane & 15)] = (_Float16)(o2[i] * inv);
        Eh[row + 48 + (lane & 15)] = (_Float16)(o3[i] * inv);
    }
}

// ---------------------------------------------------------------------------
// GEMM2: out[m,f] = E[m,:] . W0[f,:]  (M=4096, N=1024, K=1024), fp32 stores.
// ---------------------------------------------------------------------------
__global__ void __launch_bounds__(128)
gemm_out(const _Float16* __restrict__ Eh, const _Float16* __restrict__ W0h,
         float* __restrict__ out) {
    const int lane = threadIdx.x & 31;
    const int wave = threadIdx.x >> 5;
    const int tile = blockIdx.x * 4 + wave;          // 128 x 32 = 4096 tiles
    const int mt = (tile / 32) * 32;
    const int ft = (tile % 32) * 32;

    v8f c00 = {}, c01 = {}, c10 = {}, c11 = {};
#pragma unroll 2
    for (int k = 0; k < E_DIM; k += 32) {
        v16h a0 = load_a_frag(Eh, E_DIM, mt,      k, lane);
        v16h a1 = load_a_frag(Eh, E_DIM, mt + 16, k, lane);
        v16h b0 = load_bT_frag(W0h, E_DIM, ft,      k, lane);
        v16h b1 = load_bT_frag(W0h, E_DIM, ft + 16, k, lane);
        c00 = wmma_f16(a0, b0, c00);
        c01 = wmma_f16(a0, b1, c01);
        c10 = wmma_f16(a1, b0, c10);
        c11 = wmma_f16(a1, b1, c11);
    }

    const int half8 = (lane & 16) ? 8 : 0;
#pragma unroll
    for (int mi = 0; mi < 2; ++mi) {
#pragma unroll
        for (int ni = 0; ni < 2; ++ni) {
            v8f c = mi ? (ni ? c11 : c10) : (ni ? c01 : c00);
            int f = ft + ni * 16 + (lane & 15);
#pragma unroll
            for (int i = 0; i < 8; ++i) {
                int r = mt + mi * 16 + half8 + i;
                out[(size_t)r * E_DIM + f] = c[i];
            }
        }
    }
}

// ---------------------------------------------------------------------------
extern "C" void kernel_launch(void* const* d_in, const int* in_sizes, int n_in,
                              void* d_out, int out_size, void* d_ws, size_t ws_size,
                              hipStream_t stream) {
    const float* x     = (const float*)d_in[0];   // [2,2048,1024]
    const float* w_qkv = (const float*)d_in[1];   // [3072,1024]
    const float* w0    = (const float*)d_in[2];   // [1024,1024]
    float*       out   = (float*)d_out;           // [2,2048,1024]

    // workspace carve (all f16, 16B-aligned sizes)
    char* ws = (char*)d_ws;
    _Float16* Xh    = (_Float16*)ws;                 ws += (size_t)NTOK * E_DIM * 2;      // 8 MiB
    _Float16* Wqkvh = (_Float16*)ws;                 ws += (size_t)3 * E_DIM * E_DIM * 2; // 6 MiB
    _Float16* W0h   = (_Float16*)ws;                 ws += (size_t)E_DIM * E_DIM * 2;     // 2 MiB
    _Float16* Qh    = (_Float16*)ws;                 ws += (size_t)NTOK * E_DIM * 2;      // 8 MiB
    _Float16* Kh    = (_Float16*)ws;                 ws += (size_t)NTOK * E_DIM * 2;      // 8 MiB
    _Float16* Vt    = (_Float16*)ws;                 ws += (size_t)NTOK * E_DIM * 2;      // 8 MiB
    _Float16* Eh    = (_Float16*)ws;                                                     // 8 MiB

    cvt_f32_f16<<<1024, 256, 0, stream>>>(x,     Xh,    NTOK * E_DIM);
    cvt_f32_f16<<<1024, 256, 0, stream>>>(w_qkv, Wqkvh, 3 * E_DIM * E_DIM);
    cvt_f32_f16<<<512,  256, 0, stream>>>(w0,    W0h,   E_DIM * E_DIM);

    // QKV projection: 12288 32x32 wave tiles / 4 waves per block
    gemm_qkv<<<3072, 128, 0, stream>>>(Xh, Wqkvh, Qh, Kh, Vt);

    // flash attention: grid (S/64 query blocks, B*H heads), 4 waves/block
    attn_flash<<<dim3(S_LEN / 64, 32), 128, 0, stream>>>(Qh, Kh, Vt, Eh);

    // output projection: 4096 tiles / 4 waves per block
    gemm_out<<<1024, 128, 0, stream>>>(Eh, W0h, out);
}
